// MoEFFN_54623394070835
// MI455X (gfx1250) — matrix operand order repaired and checked by reading, not compiled, not run on hardware
//
#include <hip/hip_runtime.h>
#include <hip/hip_bf16.h>

#define T_TOK 2048
#define DDIM  1024
#define HDIM  4096
#define NEXP  8

typedef __attribute__((ext_vector_type(16))) __bf16 v16bf;
typedef __attribute__((ext_vector_type(8)))  float  v8f;
typedef __attribute__((ext_vector_type(4)))  int    v4i;

// ---- CDNA5 async global->LDS copy (guarded; falls back to b128 VGPR copy) --
#if defined(__HIP_DEVICE_COMPILE__) &&                                   \
    __has_builtin(__builtin_amdgcn_global_load_async_to_lds_b128) &&     \
    __has_builtin(__builtin_amdgcn_s_wait_asynccnt)
#define MOE_USE_ASYNC 1
#else
#define MOE_USE_ASYNC 0
#endif

__device__ __forceinline__ void copy16B_g2s(const void* g, void* s) {
#if MOE_USE_ASYNC
    __builtin_amdgcn_global_load_async_to_lds_b128((v4i*)g, (v4i*)s, 0, 0);
#else
    *(uint4*)s = *(const uint4*)g;
#endif
}

__device__ __forceinline__ void async_wait_all() {
#if MOE_USE_ASYNC
    __builtin_amdgcn_s_wait_asynccnt(0);
#endif
}

// round-to-nearest-even f32 -> bf16
__device__ __forceinline__ unsigned short f2bf(float f) {
    unsigned u = __float_as_uint(f);
    unsigned r = u + 0x7FFFu + ((u >> 16) & 1u);
    return (unsigned short)(r >> 16);
}

__device__ __forceinline__ float gelu_exact(float x) {
    return 0.5f * x * (1.0f + erff(x * 0.70710678118654752440f));
}

// ---------------------------------------------------------------------------
// Router: one wave per token
// ---------------------------------------------------------------------------
__global__ __launch_bounds__(256)
void moe_router_kernel(const float* __restrict__ x,
                       const float* __restrict__ rw,
                       const float* __restrict__ rb,
                       float* __restrict__ probs, int Tt) {
    const int lane = threadIdx.x & 31;
    const int wave = threadIdx.x >> 5;
    const int t = blockIdx.x * 8 + wave;
    if (t >= Tt) return;

    float acc[NEXP];
#pragma unroll
    for (int e = 0; e < NEXP; ++e) acc[e] = 0.0f;

    const float* xr = x + (size_t)t * DDIM;
    for (int d = lane; d < DDIM; d += 32) {
        float xv = xr[d];
        const float* w = rw + (size_t)d * NEXP;
#pragma unroll
        for (int e = 0; e < NEXP; ++e) acc[e] += xv * w[e];
    }
#pragma unroll
    for (int e = 0; e < NEXP; ++e) {
#pragma unroll
        for (int off = 16; off >= 1; off >>= 1)
            acc[e] += __shfl_xor(acc[e], off, 32);
    }
    if (lane == 0) {
        float mx = -3.0e38f;
#pragma unroll
        for (int e = 0; e < NEXP; ++e) { acc[e] += rb[e]; mx = fmaxf(mx, acc[e]); }
        float s = 0.0f;
#pragma unroll
        for (int e = 0; e < NEXP; ++e) { acc[e] = expf(acc[e] - mx); s += acc[e]; }
        float inv = 1.0f / s;
#pragma unroll
        for (int e = 0; e < NEXP; ++e) probs[(size_t)t * NEXP + e] = acc[e] * inv;
    }
}

// ---------------------------------------------------------------------------
// x: f32 -> bf16
// ---------------------------------------------------------------------------
__global__ __launch_bounds__(256)
void moe_cvt_bf16_kernel(const float* __restrict__ in,
                         unsigned short* __restrict__ out, int n) {
    int i = blockIdx.x * blockDim.x + threadIdx.x;
    int stride = gridDim.x * blockDim.x;
    for (; i < n; i += stride) out[i] = f2bf(in[i]);
}

// ---------------------------------------------------------------------------
// Weight convert + transpose: W[K,N] f32  ->  Wt[N,K] bf16   (64x64 LDS tiles)
// ---------------------------------------------------------------------------
__global__ __launch_bounds__(256)
void moe_w_cvt_tr_kernel(const float* __restrict__ W,
                         unsigned short* __restrict__ Wt,
                         int K, int N) {
    __shared__ unsigned short t[64][72];   // 144B rows: b128-aligned, few conflicts
    const int tid = threadIdx.x;
    const int k0 = blockIdx.y * 64, n0 = blockIdx.x * 64;
    {
        const int k  = tid >> 2;           // 0..63
        const int nc = (tid & 3) * 16;     // 0,16,32,48
        const float* gp = W + (size_t)(k0 + k) * N + n0 + nc;
#pragma unroll
        for (int j = 0; j < 16; ++j)
            t[nc + j][k] = f2bf(gp[j]);
    }
    __syncthreads();
    {
        const int n  = tid >> 2;
        const int kc = (tid & 3) * 16;
        const uint4* sp = (const uint4*)&t[n][kc];
        uint4* op = (uint4*)(Wt + (size_t)(n0 + n) * K + k0 + kc);
        op[0] = sp[0];
        op[1] = sp[1];
    }
}

// ---------------------------------------------------------------------------
// GEMM tile engine: C[M,N] (f32 acc) = A[M,K](bf16) * Bt[N,K](bf16)^T
//   block tile 128x128, K-step 32, double-buffered LDS, 8 waves (4M x 2N),
//   each wave 32x64 -> 2x4 v_wmma_f32_16x16x32_bf16 per K-step.
// ---------------------------------------------------------------------------
#define LS_STRIDE 40   // ushorts per LDS row (80B: b128-aligned, conflict-free)
#define LS_DW     20

struct WmmaAcc { v8f c[2][4]; };

__device__ __forceinline__ void tile_copy_g2s(const unsigned short* __restrict__ g,
                                              int ld, int row0, int k0,
                                              unsigned short* s, int tid) {
    const int r = tid >> 1;             // 0..127
    const int h = (tid & 1) * 16;       // half-row
    const unsigned short* gp = g + (size_t)(row0 + r) * ld + k0 + h;
    unsigned short* sp = s + r * LS_STRIDE + h;
    copy16B_g2s(gp,     sp);
    copy16B_g2s(gp + 8, sp + 8);
}

__device__ __forceinline__ void gemm_tile_loop(
    const unsigned short* __restrict__ Ab,   // [M,K] bf16
    const unsigned short* __restrict__ Bt,   // [N,K] bf16
    unsigned short* lsA, unsigned short* lsB,   // each 2 * 128*LS_STRIDE
    int m0, int n0, int K,
    int tid, int lane, int wm, int wn, WmmaAcc& A)
{
#pragma unroll
    for (int mb = 0; mb < 2; ++mb)
#pragma unroll
        for (int nb = 0; nb < 4; ++nb)
#pragma unroll
            for (int i = 0; i < 8; ++i) A.c[mb][nb][i] = 0.0f;

    const int BUF = 128 * LS_STRIDE;
    const int kq = lane >> 4;
    const int ln = lane & 15;
    const int nk = K >> 5;

    // prologue: stage tile 0 into buffer 0
    tile_copy_g2s(Ab, K, m0, 0, lsA, tid);
    tile_copy_g2s(Bt, K, n0, 0, lsB, tid);
    async_wait_all();
    __syncthreads();

    for (int kb = 0; kb < nk; ++kb) {
        const int p = kb & 1;
        // stage next tile into the other buffer (overlaps with WMMA below)
        if (kb + 1 < nk) {
            tile_copy_g2s(Ab, K, m0, (kb + 1) << 5, lsA + (p ^ 1) * BUF, tid);
            tile_copy_g2s(Bt, K, n0, (kb + 1) << 5, lsB + (p ^ 1) * BUF, tid);
        }
        const unsigned* a32 = (const unsigned*)(lsA + p * BUF);
        const unsigned* b32 = (const unsigned*)(lsB + p * BUF);

        union Frag { unsigned u[8]; v16bf v; };
        Frag af[2], bfr[4];
#pragma unroll
        for (int mb = 0; mb < 2; ++mb) {
            const int row = wm * 32 + mb * 16 + ln;
#pragma unroll
            for (int i = 0; i < 4; ++i) {
                af[mb].u[i]     = a32[row * LS_DW + kq * 8 + i];
                af[mb].u[4 + i] = a32[row * LS_DW + kq * 8 + 4 + i];
            }
        }
#pragma unroll
        for (int nb = 0; nb < 4; ++nb) {
            const int col = wn * 64 + nb * 16 + ln;
#pragma unroll
            for (int i = 0; i < 8; ++i)
                bfr[nb].u[i] = b32[col * LS_DW + kq * 8 + i];
        }
#pragma unroll
        for (int mb = 0; mb < 2; ++mb)
#pragma unroll
            for (int nb = 0; nb < 4; ++nb)
                A.c[mb][nb] = __builtin_amdgcn_wmma_f32_16x16x32_bf16(
                    false, af[mb].v, false, bfr[nb].v,
                    (short)0, A.c[mb][nb], false, false);

        async_wait_all();      // next tile fully landed in LDS
        __syncthreads();       // one barrier per K-step
    }
}

// ---------------------------------------------------------------------------
// FC1 + exact GELU -> bf16 h
// ---------------------------------------------------------------------------
__global__ __launch_bounds__(256)
void moe_fc1_gelu_kernel(const unsigned short* __restrict__ Xb,
                         const unsigned short* __restrict__ W1t,  // [H,D] bf16
                         const float* __restrict__ b1,
                         unsigned short* __restrict__ Hb,
                         int M, int N, int K)
{
    __shared__ unsigned short lsA[2 * 128 * LS_STRIDE];
    __shared__ unsigned short lsB[2 * 128 * LS_STRIDE];
    const int tid = threadIdx.x;
    const int lane = tid & 31, wave = tid >> 5;
    const int wm = wave & 3, wn = wave >> 2;
    const int m0 = blockIdx.y * 128, n0 = blockIdx.x * 128;

    WmmaAcc A;
    gemm_tile_loop(Xb, W1t, lsA, lsB, m0, n0, K, tid, lane, wm, wn, A);

    const int ln = lane & 15, kq = lane >> 4;
#pragma unroll
    for (int mb = 0; mb < 2; ++mb) {
#pragma unroll
        for (int nb = 0; nb < 4; ++nb) {
            const int col = n0 + wn * 64 + nb * 16 + ln;
            const float bv = b1[col];
#pragma unroll
            for (int i = 0; i < 8; ++i) {
                const int row = m0 + wm * 32 + mb * 16 + kq * 8 + i;
                float v = A.c[mb][nb][i] + bv;
                Hb[(size_t)row * N + col] = f2bf(gelu_exact(v));
            }
        }
    }
}

// ---------------------------------------------------------------------------
// FC2 + probs-scale, accumulate into y (expert 0 initializes)
// ---------------------------------------------------------------------------
__global__ __launch_bounds__(256)
void moe_fc2_acc_kernel(const unsigned short* __restrict__ Hb,
                        const unsigned short* __restrict__ W2t,  // [D,H] bf16
                        const float* __restrict__ b2,
                        const float* __restrict__ probs,
                        int expert, int initFlag,
                        float* __restrict__ Y,
                        int M, int N, int K)
{
    __shared__ unsigned short lsA[2 * 128 * LS_STRIDE];
    __shared__ unsigned short lsB[2 * 128 * LS_STRIDE];
    const int tid = threadIdx.x;
    const int lane = tid & 31, wave = tid >> 5;
    const int wm = wave & 3, wn = wave >> 2;
    const int m0 = blockIdx.y * 128, n0 = blockIdx.x * 128;

    WmmaAcc A;
    gemm_tile_loop(Hb, W2t, lsA, lsB, m0, n0, K, tid, lane, wm, wn, A);

    const int ln = lane & 15, kq = lane >> 4;
#pragma unroll
    for (int mb = 0; mb < 2; ++mb) {
        float pr[8];
#pragma unroll
        for (int i = 0; i < 8; ++i) {
            const int row = m0 + wm * 32 + mb * 16 + kq * 8 + i;
            pr[i] = probs[(size_t)row * NEXP + expert];
        }
#pragma unroll
        for (int nb = 0; nb < 4; ++nb) {
            const int col = n0 + wn * 64 + nb * 16 + ln;
            const float bv = b2[col];
#pragma unroll
            for (int i = 0; i < 8; ++i) {
                const int row = m0 + wm * 32 + mb * 16 + kq * 8 + i;
                const float v = (A.c[mb][nb][i] + bv) * pr[i];
                const size_t idx = (size_t)row * N + col;
                if (initFlag) Y[idx] = v;
                else          Y[idx] += v;
            }
        }
    }
}

// ---------------------------------------------------------------------------
extern "C" void kernel_launch(void* const* d_in, const int* in_sizes, int n_in,
                              void* d_out, int out_size, void* d_ws, size_t ws_size,
                              hipStream_t stream) {
    (void)in_sizes; (void)n_in; (void)out_size; (void)ws_size;
    const float* x   = (const float*)d_in[0];
    const float* rw  = (const float*)d_in[1];
    const float* rb  = (const float*)d_in[2];
    const float* w1  = (const float*)d_in[3];
    const float* b1  = (const float*)d_in[4];
    const float* w2  = (const float*)d_in[5];
    const float* b2  = (const float*)d_in[6];

    float* y     = (float*)d_out;                        // [T, D]
    float* probs = y + (size_t)T_TOK * DDIM;             // [T, E]

    unsigned short* xb  = (unsigned short*)d_ws;         // [T, D]  bf16  (4 MB)
    unsigned short* hb  = xb  + (size_t)T_TOK * DDIM;    // [T, H]  bf16  (16 MB)
    unsigned short* w1t = hb  + (size_t)T_TOK * HDIM;    // [H, D]  bf16  (8 MB)
    unsigned short* w2t = w1t + (size_t)HDIM * DDIM;     // [D, H]  bf16  (8 MB)

    moe_router_kernel<<<T_TOK / 8, 256, 0, stream>>>(x, rw, rb, probs, T_TOK);
    moe_cvt_bf16_kernel<<<2048, 256, 0, stream>>>(x, xb, T_TOK * DDIM);

    for (int e = 0; e < NEXP; ++e) {
        // convert+transpose this expert's weights to bf16 [N,K]
        moe_w_cvt_tr_kernel<<<dim3(HDIM / 64, DDIM / 64), 256, 0, stream>>>(
            w1 + (size_t)e * DDIM * HDIM, w1t, DDIM, HDIM);
        moe_fc1_gelu_kernel<<<dim3(HDIM / 128, T_TOK / 128), 256, 0, stream>>>(
            xb, w1t, b1 + (size_t)e * HDIM, hb, T_TOK, HDIM, DDIM);

        moe_w_cvt_tr_kernel<<<dim3(DDIM / 64, HDIM / 64), 256, 0, stream>>>(
            w2 + (size_t)e * HDIM * DDIM, w2t, HDIM, DDIM);
        moe_fc2_acc_kernel<<<dim3(DDIM / 128, T_TOK / 128), 256, 0, stream>>>(
            hb, w2t, b2 + (size_t)e * DDIM, probs,
            e, (e == 0) ? 1 : 0, y, T_TOK, DDIM, HDIM);
    }
}
